// INT8Linear_62285615726858
// MI455X (gfx1250) — 8-bit matrix operand order repaired and checked by reading, compiled, not run on hardware
//
#include <hip/hip_runtime.h>

typedef __attribute__((ext_vector_type(8))) int v8i;

#define T_TOK 8192
#define K_DIM 4096
#define O_DIM 11008
#define KB    32
#define QMAXF 127.0f

constexpr int LSTR = 144;

__global__ __launch_bounds__(256) void quant_act_kernel(
    const float* __restrict__ x, signed char* __restrict__ xq,
    float* __restrict__ sa)
{
    int lane = threadIdx.x & 31;
    int wave = threadIdx.x >> 5;
    int g  = blockIdx.x * 8 + wave;
    int t  = g >> 5;
    int kb = g & 31;

    const float* p = x + (size_t)t * K_DIM + kb * 128 + lane * 4;
    float4 v = *(const float4*)p;
    float amax = fmaxf(fmaxf(fabsf(v.x), fabsf(v.y)),
                       fmaxf(fabsf(v.z), fabsf(v.w)));
    for (int off = 16; off; off >>= 1)
        amax = fmaxf(amax, __shfl_xor(amax, off));

    float scale = (amax > 0.f) ? amax * (1.0f / QMAXF) : 1.0f;
    float inv   = (amax > 0.f) ? QMAXF / amax : 1.0f;

    int q0 = (int)rintf(v.x * inv); q0 = min(127, max(-127, q0));
    int q1 = (int)rintf(v.y * inv); q1 = min(127, max(-127, q1));
    int q2 = (int)rintf(v.z * inv); q2 = min(127, max(-127, q2));
    int q3 = (int)rintf(v.w * inv); q3 = min(127, max(-127, q3));
    unsigned packed = (unsigned)(q0 & 255) | ((unsigned)(q1 & 255) << 8) |
                      ((unsigned)(q2 & 255) << 16) | ((unsigned)(q3 & 255) << 24);

    *(unsigned*)(xq + (size_t)t * K_DIM + kb * 128 + lane * 4) = packed;
    if (lane == 0) sa[t * KB + kb] = scale;
}

__global__ __launch_bounds__(256) void quant_w_kernel(
    const float* __restrict__ w, signed char* __restrict__ wq,
    float* __restrict__ sw)
{
    __shared__ float red[256];
    int ob  = blockIdx.y;
    int kb  = blockIdx.x;
    int tid = threadIdx.x;
    int row = tid >> 1;
    int ch  = (tid & 1) * 64;

    const float* base = w + (size_t)(ob * 128 + row) * K_DIM + kb * 128 + ch;
    float amax = 0.f;
    for (int j = 0; j < 16; ++j) {
        float4 v = ((const float4*)base)[j];
        amax = fmaxf(amax, fmaxf(fmaxf(fabsf(v.x), fabsf(v.y)),
                                 fmaxf(fabsf(v.z), fabsf(v.w))));
    }
    red[tid] = amax;
    __syncthreads();
    for (int s = 128; s; s >>= 1) {
        if (tid < s) red[tid] = fmaxf(red[tid], red[tid + s]);
        __syncthreads();
    }
    float am    = red[0];
    float scale = (am > 0.f) ? am * (1.0f / QMAXF) : 1.0f;
    float inv   = (am > 0.f) ? QMAXF / am : 1.0f;
    if (tid == 0) sw[ob * KB + kb] = scale;

    unsigned* qb = (unsigned*)(wq + (size_t)(ob * 128 + row) * K_DIM + kb * 128 + ch);
    for (int j = 0; j < 16; ++j) {
        float4 v = ((const float4*)base)[j];
        int q0 = (int)rintf(v.x * inv); q0 = min(127, max(-127, q0));
        int q1 = (int)rintf(v.y * inv); q1 = min(127, max(-127, q1));
        int q2 = (int)rintf(v.z * inv); q2 = min(127, max(-127, q2));
        int q3 = (int)rintf(v.w * inv); q3 = min(127, max(-127, q3));
        qb[j] = (unsigned)(q0 & 255) | ((unsigned)(q1 & 255) << 8) |
                ((unsigned)(q2 & 255) << 16) | ((unsigned)(q3 & 255) << 24);
    }
}

__device__ __forceinline__ void async_copy_tiles(
    const signed char* __restrict__ gA, const signed char* __restrict__ gB,
    signed char* sA, signed char* sB, int tid)
{
    for (int i = 0; i < 4; ++i) {
        int lin = tid + i * 256;
        int r = lin >> 3;
        int c = (lin & 7) * 16;
        unsigned la = (unsigned)(unsigned long long)(sA + r * LSTR + c);
        const signed char* ga = gA + (size_t)r * K_DIM + c;
        asm volatile("global_load_async_to_lds_b128 %0, %1, off"
                     :: "v"(la), "v"(ga) : "memory");
        unsigned lb = (unsigned)(unsigned long long)(sB + r * LSTR + c);
        const signed char* gb = gB + (size_t)r * K_DIM + c;
        asm volatile("global_load_async_to_lds_b128 %0, %1, off"
                     :: "v"(lb), "v"(gb) : "memory");
    }
}

__global__ __launch_bounds__(256) void gemm_i8_kernel(
    const signed char* __restrict__ xq, const float* __restrict__ sa,
    const signed char* __restrict__ wq, const float* __restrict__ sw,
    const float* __restrict__ bias, float* __restrict__ out)
{
    __shared__ alignas(16) signed char lA[2][128 * LSTR];
    __shared__ alignas(16) signed char lB[2][128 * LSTR];
    __shared__ float lSA[128 * KB];

    int tid   = threadIdx.x;
    int lane  = tid & 31;
    int wave  = tid >> 5;
    int waveM = wave >> 2;
    int waveN = wave & 3;
    int tileM = blockIdx.y;
    int tileN = blockIdx.x;
    int hi    = (lane & 16) ? 1 : 0;
    int l15   = lane & 15;

    const signed char* aG = xq + (size_t)tileM * 128 * K_DIM;
    const signed char* bG = wq + (size_t)tileN * 128 * K_DIM;

    async_copy_tiles(aG, bG, lA[0], lB[0], tid);

    for (int i = 0; i < 16; ++i) {
        int idx = tid + i * 256;
        int r = idx >> 5, k = idx & 31;
        lSA[idx] = sa[(size_t)(tileM * 128 + r) * KB + k];
    }

    float acc[4][2][8];
    for (int mi = 0; mi < 4; ++mi)
        for (int ni = 0; ni < 2; ++ni)
            for (int e = 0; e < 8; ++e) acc[mi][ni][e] = 0.f;

    for (int kb = 0; kb < KB; ++kb) {
        int cur = kb & 1;
        if (kb + 1 < KB) {
            async_copy_tiles(aG + (size_t)(kb + 1) * 128,
                             bG + (size_t)(kb + 1) * 128,
                             lA[cur ^ 1], lB[cur ^ 1], tid);
            asm volatile("s_wait_asynccnt 0x8" ::: "memory");
        } else {
            asm volatile("s_wait_asynccnt 0x0" ::: "memory");
        }
        __syncthreads();

        float swk = sw[tileN * KB + kb];

        v8i bf[2][2];
        for (int ni = 0; ni < 2; ++ni) {
            int n = waveN * 32 + ni * 16 + l15;
            const signed char* bp = lB[cur] + n * LSTR + hi * 16;
            for (int h = 0; h < 2; ++h) {
                int4 lo = *(const int4*)(bp + h * 64);
                int4 up = *(const int4*)(bp + h * 64 + 32);
                v8i b;
                b[0] = lo.x; b[1] = lo.y; b[2] = lo.z; b[3] = lo.w;
                b[4] = up.x; b[5] = up.y; b[6] = up.z; b[7] = up.w;
                bf[ni][h] = b;
            }
        }

        for (int mi = 0; mi < 4; ++mi) {
            int mrow = waveM * 64 + mi * 16;
            int m = mrow + l15;
            const signed char* ap = lA[cur] + m * LSTR + hi * 16;

            v8i af[2];
            for (int h = 0; h < 2; ++h) {
                v8i a;
                for (int p = 0; p < 4; ++p) {
                    int2 d = *(const int2*)(ap + h * 64 + p * 8 + (p >> 1) * 16);
                    a[2 * p]     = d.x;
                    a[2 * p + 1] = d.y;
                }
                af[h] = a;
            }

            float srow[8];
            for (int e = 0; e < 8; ++e)
                srow[e] = lSA[(mrow + e + hi * 8) * KB + kb] * swk;

            for (int ni = 0; ni < 2; ++ni) {
                v8i c = {0, 0, 0, 0, 0, 0, 0, 0};
                c = __builtin_amdgcn_wmma_i32_16x16x64_iu8(
                        true, af[0], true, bf[ni][0], c, false, false);
                c = __builtin_amdgcn_wmma_i32_16x16x64_iu8(
                        true, af[1], true, bf[ni][1], c, false, false);
                for (int e = 0; e < 8; ++e)
                    acc[mi][ni][e] += (float)c[e] * srow[e];
            }
        }
        __syncthreads();
    }

    for (int ni = 0; ni < 2; ++ni) {
        int col = tileN * 128 + waveN * 32 + ni * 16 + l15;
        float bz = bias[col];
        for (int mi = 0; mi < 4; ++mi) {
            int rbase = tileM * 128 + waveM * 64 + mi * 16 + hi * 8;
            for (int e = 0; e < 8; ++e)
                out[(size_t)(rbase + e) * O_DIM + col] = acc[mi][ni][e] + bz;
        }
    }
}

extern "C" void kernel_launch(void* const* d_in, const int* in_sizes, int n_in,
                              void* d_out, int out_size, void* d_ws, size_t ws_size,
                              hipStream_t stream) {
    const float* x    = (const float*)d_in[0];
    const float* w    = (const float*)d_in[1];
    const float* bias = (const float*)d_in[2];
    float* out        = (float*)d_out;

    char* ws = (char*)d_ws;
    signed char* xq = (signed char*)ws;
    signed char* wq = (signed char*)(ws + (size_t)33554432);
    float* sa = (float*)(ws + (size_t)33554432 + 45088768);
    float* sw = (float*)(ws + (size_t)33554432 + 45088768 + 1048576);

    quant_act_kernel<<<dim3((T_TOK * KB) / 8), 256, 0, stream>>>(x, xq, sa);
    quant_w_kernel<<<dim3(KB, O_DIM / 128), 256, 0, stream>>>(w, wq, sw);
    gemm_i8_kernel<<<dim3(O_DIM / 128, T_TOK / 128), 256, 0, stream>>>(
        xq, sa, wq, sw, bias, out);
}
